// PGnet_88373247082886
// MI455X (gfx1250) — matrix operand order repaired and checked by reading, not compiled, not run on hardware
//
// PGnet for MI455X (gfx1250, wave32, WGP, WMMA).
//
// Roofline: fpg_w + rpg_w = 2 x 602 MB dominate. Strategy:
//   K1 (phase1): 16 batch elems / workgroup.
//     - stream fpg_w ONCE through LDS staging -> M_f[b,16,14] (inp folded in)
//     - cache rpg_w tile in LDS as f16 (293,888 B) -> 5-step recurrence runs
//       entirely out of LDS; rw read from HBM exactly once here
//     - ctrl GEMM [16 x 98] @ W^T via V_WMMA_F32_16X16X32_F16 (K padded to 128,
//       4 chunks), wave 0 per workgroup; B fragments held in VGPRs
//     - emits final pre-softmax logits z[B,14] and M_f to workspace
//   K2: per-column (14) max / sum-exp over the batch axis (softmax axis=0)
//   K3: final differentiable pass; re-reads rpg_w once in f32 (coalesced
//       float4, LDS f32 atomics for the 41 outputs), writes l1i/l2/rc.
// HBM traffic ~ 1.9 GB -> ~80 us at 23.3 TB/s (vs 7.2 GB naive).

#include <hip/hip_runtime.h>

#define BSZ     16384
#define INDIM   41
#define L2DIM   16
#define RCDIM   14
#define CTRLK   128          // 98 padded to 128 for f16 WMMA (4 x K=32)
#define PER_B   9184         // 16*41*14 == 41*16*14 floats per batch element
#define BT      16           // batch tile per workgroup (matches WMMA M)

typedef _Float16 v16h __attribute__((ext_vector_type(16)));
typedef float    v8f  __attribute__((ext_vector_type(8)));

__device__ __forceinline__ float clipf(float v, float lo, float hi) {
  return fminf(fmaxf(v, lo), hi);
}

// ---------------------------------------------------------------------------
// Kernel 1: recurrent phase. 1024 workgroups x 256 threads, 322,112 B dyn LDS.
// ---------------------------------------------------------------------------
__global__ __launch_bounds__(256, 1)
void pg_phase1(const float* __restrict__ inp,  const float* __restrict__ pgw,
               const float* __restrict__ pgb,  const float* __restrict__ fpw,
               const float* __restrict__ fpb,  const float* __restrict__ rpw,
               const float* __restrict__ rpb,  const float* __restrict__ l2in,
               const float* __restrict__ l1in, float* __restrict__ mf_ws,
               float* __restrict__ zbuf)
{
  extern __shared__ char smem[];
  _Float16* rwlds = (_Float16*)smem;                  // 293,888 B (also f32 staging)
  float* xbuf  = (float*)(smem + (size_t)BT * PER_B * 2); // [16][128]
  float* Mf    = xbuf  + BT * CTRLK;                  // [16][224]
  float* l1i_s = Mf    + BT * 224;                    // [16][41]
  float* l2_s  = l1i_s + BT * INDIM;                  // [16][16]
  float* rc_s  = l2_s  + BT * L2DIM;                  // [16][16]
  float* tmpA  = rc_s  + BT * L2DIM;                  // [16][16] scratch

  const int tid = threadIdx.x;
  const int b0  = blockIdx.x * BT;

  // ---- init x = [inp | l2_init | inp - l1i_init | 0-pad], plus l2/l1i state
  for (int idx = tid; idx < BT * CTRLK; idx += 256) {
    int b = idx >> 7, k = idx & 127;
    float v;
    if (k < 41)      v = inp[(size_t)(b0 + b) * INDIM + k];
    else if (k < 57) v = l2in[(size_t)(b0 + b) * L2DIM + (k - 41)];
    else if (k < 98) v = inp[(size_t)(b0 + b) * INDIM + (k - 57)]
                       - l1in[(size_t)(b0 + b) * INDIM + (k - 57)];
    else             v = 0.0f;
    xbuf[idx] = v;
  }
  for (int idx = tid; idx < BT * L2DIM; idx += 256)
    l2_s[idx] = l2in[(size_t)b0 * L2DIM + idx];
  for (int idx = tid; idx < BT * INDIM; idx += 256)
    l1i_s[idx] = l1in[(size_t)b0 * INDIM + idx];

  // ---- B fragments (ctrl weights) resident in wave-0 VGPRs.
  // B (32x16 f16) layout chosen here (contiguous k, matching A): lane L -> col N = L&15; elem e -> K = e + 16*(L>=16).
  v16h bfrag[4];
  if (tid < 32) {
    const int j = tid & 15, koff = (tid >= 16) ? 16 : 0;
    for (int c = 0; c < 4; ++c) {
      v16h f;
#pragma unroll
      for (int e = 0; e < 16; ++e) {
        int k = c * 32 + koff + e;
        float w = (j < RCDIM && k < 98) ? pgw[j * 98 + k] : 0.0f;
        f[e] = (_Float16)w;
      }
      bfrag[c] = f;
    }
  }
  __syncthreads();

  // ---- M_f[b][o][r] = sum_i clip(fw[b,o,i,r]) * inp[b,i]; fw read ONCE.
  // Stage 8 batch elems of fw (293,888 B) through the rw LDS region at a time.
  {
    float* stage = (float*)rwlds;
    for (int h = 0; h < 2; ++h) {
      const float4* src = (const float4*)(fpw + (size_t)(b0 + 8 * h) * PER_B);
      for (int idx = tid; idx < 8 * PER_B / 4; idx += 256)
        ((float4*)stage)[idx] = src[idx];
      __syncthreads();
      for (int idx = tid; idx < 8 * 224; idx += 256) {
        int bl = idx / 224, orr = idx % 224, o = orr / 14, r = orr % 14;
        const float* wb = stage + bl * PER_B + o * (INDIM * RCDIM) + r;
        const float* xr = xbuf + (8 * h + bl) * CTRLK;
        float acc = 0.0f;
        for (int i = 0; i < INDIM; ++i)
          acc += clipf(wb[i * RCDIM], -1.0f, 1.0f) * xr[i];
        Mf[(8 * h + bl) * 224 + orr] = acc;
      }
      __syncthreads();
    }
  }

  // ---- load rw tile once: f32 global (float4, coalesced) -> f16 LDS (b64 stores)
  {
    const float4* src = (const float4*)(rpw + (size_t)b0 * PER_B);
    for (int idx = tid; idx < BT * PER_B / 4; idx += 256) {
      float4 v = src[idx];
      union { _Float16 h[4]; uint2 u; } pk;
      pk.h[0] = (_Float16)clipf(v.x, -1.0f, 1.0f);
      pk.h[1] = (_Float16)clipf(v.y, -1.0f, 1.0f);
      pk.h[2] = (_Float16)clipf(v.z, -1.0f, 1.0f);
      pk.h[3] = (_Float16)clipf(v.w, -1.0f, 1.0f);
      *(uint2*)(rwlds + (size_t)idx * 4) = pk.u;
    }
  }
  __syncthreads();

  // ---- recurrence: t=0..4 full steps; t=5 computes final ctrl logits only
  for (int t = 0; t < 6; ++t) {
    // refresh x cols 41..97 from current l2 / l1i
    for (int idx = tid; idx < BT * 57; idx += 256) {
      int b = idx / 57, c = idx % 57;
      if (c < 16) xbuf[b * CTRLK + 41 + c] = l2_s[b * L2DIM + c];
      else {
        int i = c - 16;
        xbuf[b * CTRLK + 57 + i] = xbuf[b * CTRLK + i] - l1i_s[b * INDIM + i];
      }
    }
    __syncthreads();

    // ctrl GEMM on wave 0: C[16b x 16j] = A[16 x 128] * B[128 x 16]
    if (tid < 32) {
      const int m = tid & 15, ah = (tid >= 16) ? 16 : 0;
      v8f acc = {0.f, 0.f, 0.f, 0.f, 0.f, 0.f, 0.f, 0.f};
#pragma unroll
      for (int c = 0; c < 4; ++c) {
        // A (16x32 f16) layout chosen here (contiguous k, matching B): lane L -> row M = L&15;
        // elem e -> K = e + 16*(L>=16)
        v16h a;
#pragma unroll
        for (int e = 0; e < 16; ++e) {
          int kl = ah + e;
          a[e] = (_Float16)xbuf[m * CTRLK + c * 32 + kl];
        }
        acc = __builtin_amdgcn_wmma_f32_16x16x32_f16(
            false, a, false, bfrag[c], (short)0, acc, false, false);
      }
      // C/D layout: elem v -> M = v + 8*(L>=16), N = L&15
      const int n = tid & 15, mh = (tid >= 16) ? 8 : 0;
#pragma unroll
      for (int v = 0; v < 8; ++v) tmpA[(v + mh) * 16 + n] = acc[v];
    }
    __syncthreads();

    if (t == 5) {  // final logits: leaky_relu only (no clip, no blur)
      for (int idx = tid; idx < BT * RCDIM; idx += 256) {
        int b = idx / 14, j = idx % 14;
        float s = tmpA[b * 16 + j] + pgb[j];
        zbuf[(size_t)(b0 + b) * RCDIM + j] = (s >= 0.0f) ? s : 0.2f * s;
      }
      break;
    }

    // rc = blur_rc( clip( leaky(raw + bias), -0.1, 1 ) )
    for (int idx = tid; idx < BT * RCDIM; idx += 256) {
      int b = idx / 14, j = idx % 14;
      float s = tmpA[b * 16 + j] + pgb[j];
      s = (s >= 0.0f) ? s : 0.2f * s;
      tmpA[b * 16 + j] = clipf(s, -0.1f, 1.0f);
    }
    __syncthreads();
    for (int idx = tid; idx < BT * RCDIM; idx += 256) {
      int b = idx / 14, j = idx % 14;
      float s = 0.85f * tmpA[b * 16 + j];
      if (j > 0)  s += 0.075f * tmpA[b * 16 + j - 1];
      if (j < 13) s += 0.075f * tmpA[b * 16 + j + 1];
      rc_s[b * 16 + j] = s;
    }
    __syncthreads();

    // l2 = blur_l2( clip( M_f . rc + fpg_b, -0.1, 1 ) )
    for (int idx = tid; idx < BT * L2DIM; idx += 256) {
      int b = idx >> 4, o = idx & 15;
      float acc = fpb[(size_t)(b0 + b) * L2DIM + o];
      const float* mrow = Mf + b * 224 + o * 14;
#pragma unroll
      for (int r = 0; r < RCDIM; ++r) acc += mrow[r] * rc_s[b * 16 + r];
      tmpA[idx] = clipf(acc, -0.1f, 1.0f);
    }
    __syncthreads();
    for (int idx = tid; idx < BT * L2DIM; idx += 256) {
      int o = idx & 15;
      float s = 0.85f * tmpA[idx];
      if (o > 0)  s += 0.075f * tmpA[idx - 1];
      if (o < 15) s += 0.075f * tmpA[idx + 1];
      l2_s[idx] = s;
    }
    __syncthreads();

    // l1i = clip( rw . (l2 (x) rc) + rpg_b, 0, 1 ) — rw from LDS (f16)
    for (int idx = tid; idx < BT * INDIM; idx += 256) {
      int b = idx / 41, ip = idx % 41;
      float acc = rpb[(size_t)(b0 + b) * INDIM + ip];
      const _Float16* wrow = rwlds + (size_t)b * PER_B + ip * 224;
      for (int l = 0; l < L2DIM; ++l) {
        float fl = l2_s[b * 16 + l];
        const _Float16* wl = wrow + l * 14;
#pragma unroll
        for (int r = 0; r < RCDIM; ++r)
          acc += (float)wl[r] * fl * rc_s[b * 16 + r];
      }
      l1i_s[idx] = clipf(acc, 0.0f, 1.0f);
    }
    __syncthreads();
  }

  // spill M_f for the final pass
  for (int idx = tid; idx < BT * 224; idx += 256)
    mf_ws[(size_t)b0 * 224 + idx] = Mf[idx];
}

// ---------------------------------------------------------------------------
// Kernel 2: per-column max & sum(exp) over batch axis (softmax axis=0)
// ---------------------------------------------------------------------------
__global__ __launch_bounds__(256)
void pg_softmax_stats(const float* __restrict__ zbuf, float* __restrict__ stats)
{
  __shared__ float red[256];
  const int j = blockIdx.x, tid = threadIdx.x;
  float m = -3.402823466e38f;
  for (int b = tid; b < BSZ; b += 256) m = fmaxf(m, zbuf[(size_t)b * RCDIM + j]);
  red[tid] = m; __syncthreads();
  for (int s = 128; s > 0; s >>= 1) {
    if (tid < s) red[tid] = fmaxf(red[tid], red[tid + s]);
    __syncthreads();
  }
  const float M = red[0]; __syncthreads();
  float sum = 0.0f;
  for (int b = tid; b < BSZ; b += 256) sum += expf(zbuf[(size_t)b * RCDIM + j] - M);
  red[tid] = sum; __syncthreads();
  for (int s = 128; s > 0; s >>= 1) {
    if (tid < s) red[tid] += red[tid + s];
    __syncthreads();
  }
  if (tid == 0) { stats[j] = M; stats[RCDIM + j] = red[0]; }
}

// ---------------------------------------------------------------------------
// Kernel 3: final pass. One workgroup per batch element; re-reads rw in f32.
// ---------------------------------------------------------------------------
__global__ __launch_bounds__(256)
void pg_final(const float* __restrict__ zbuf, const float* __restrict__ stats,
              const float* __restrict__ mf_ws, const float* __restrict__ fpb,
              const float* __restrict__ rpw,   const float* __restrict__ rpb,
              float* __restrict__ out_l1i, float* __restrict__ out_l2,
              float* __restrict__ out_rc)
{
  __shared__ float rcf[16];
  __shared__ float l2f[16];
  __shared__ float G[224];
  __shared__ float accI[INDIM];
  const int b = blockIdx.x, tid = threadIdx.x;

  if (tid < RCDIM) {
    float v = expf(zbuf[(size_t)b * RCDIM + tid] - stats[tid]) / stats[RCDIM + tid];
    rcf[tid] = v;
    out_rc[(size_t)b * RCDIM + tid] = v;
  }
  if (tid >= RCDIM && tid < 16) rcf[tid] = 0.0f;
  if (tid < INDIM) accI[tid] = 0.0f;
  __syncthreads();

  if (tid < L2DIM) {
    float acc = fpb[(size_t)b * L2DIM + tid];
    const float* mrow = mf_ws + (size_t)b * 224 + tid * 14;
#pragma unroll
    for (int r = 0; r < RCDIM; ++r) acc += mrow[r] * rcf[r];
    acc = clipf(acc, -0.1f, 1.0f);   // final pass: clip, NO blur
    l2f[tid] = acc;
    out_l2[(size_t)b * L2DIM + tid] = acc;
  }
  __syncthreads();

  for (int idx = tid; idx < 224; idx += 256) G[idx] = l2f[idx / 14] * rcf[idx % 14];
  __syncthreads();

  const float4* src = (const float4*)(rpw + (size_t)b * PER_B);
  for (int idx = tid; idx < PER_B / 4; idx += 256) {
    float4 w = src[idx];
    int e0 = idx * 4, ip = e0 / 224, lr = e0 % 224; // 224 % 4 == 0: same row
    float p = clipf(w.x, -1.f, 1.f) * G[lr]
            + clipf(w.y, -1.f, 1.f) * G[lr + 1]
            + clipf(w.z, -1.f, 1.f) * G[lr + 2]
            + clipf(w.w, -1.f, 1.f) * G[lr + 3];
    atomicAdd(&accI[ip], p);  // ds_add_f32
  }
  __syncthreads();

  if (tid < INDIM)
    out_l1i[(size_t)b * INDIM + tid] =
        clipf(accI[tid] + rpb[(size_t)b * INDIM + tid], 0.0f, 1.0f);
}

// ---------------------------------------------------------------------------
extern "C" void kernel_launch(void* const* d_in, const int* in_sizes, int n_in,
                              void* d_out, int out_size, void* d_ws, size_t ws_size,
                              hipStream_t stream)
{
  (void)in_sizes; (void)n_in; (void)out_size; (void)ws_size;
  const float* inp  = (const float*)d_in[0];
  const float* pgw  = (const float*)d_in[1];
  const float* pgb  = (const float*)d_in[2];
  const float* fpw  = (const float*)d_in[3];
  const float* fpb  = (const float*)d_in[4];
  const float* rpw  = (const float*)d_in[5];
  const float* rpb  = (const float*)d_in[6];
  const float* l2in = (const float*)d_in[7];
  const float* l1in = (const float*)d_in[8];

  // workspace: M_f[B,224] | z[B,14] | stats[28]  (~15.6 MB total)
  float* mf_ws = (float*)d_ws;
  float* zbuf  = mf_ws + (size_t)BSZ * 224;
  float* stats = zbuf + (size_t)BSZ * RCDIM;

  float* out_l1i = (float*)d_out;
  float* out_l2  = out_l1i + (size_t)BSZ * INDIM;
  float* out_rc  = out_l2 + (size_t)BSZ * L2DIM;

  // dyn LDS: rw f16 tile + x + M_f + states = 322,112 B (< 320 KiB/WGP)
  const size_t SMEM1 = (size_t)BT * PER_B * 2 +
                       (size_t)(BT * CTRLK + BT * 224 + BT * INDIM +
                                3 * BT * L2DIM) * sizeof(float);
  hipFuncSetAttribute(reinterpret_cast<const void*>(pg_phase1),
                      hipFuncAttributeMaxDynamicSharedMemorySize, (int)SMEM1);

  pg_phase1<<<BSZ / BT, 256, SMEM1, stream>>>(inp, pgw, pgb, fpw, fpb, rpw, rpb,
                                              l2in, l1in, mf_ws, zbuf);
  pg_softmax_stats<<<RCDIM, 256, 0, stream>>>(zbuf, stats);
  pg_final<<<BSZ, 256, 0, stream>>>(zbuf, stats, mf_ws, fpb, rpw, rpb,
                                    out_l1i, out_l2, out_rc);
}
